// MyNet_17222818857297
// MI455X (gfx1250) — compile-verified, matrix operand order checked by reading, not run on hardware
//
#include <hip/hip_runtime.h>

// CDNA5 / gfx1250: wave32, WMMA 16x16x32 f16 (codegen-confirmed builtin).

typedef __attribute__((ext_vector_type(16))) _Float16 v16h;
typedef __attribute__((ext_vector_type(8)))  float    v8f;

// One wave (32 lanes) per block. Each block redundantly computes the
// 128-element softmax (trivial compute), then writes 8 of the 128 identical
// output rows. 16 blocks x 8 rows = 128 rows, 64 KB total stores.
__global__ __launch_bounds__(32)
void softmax_conv_wmma_kernel(const float* __restrict__ x,
                              float* __restrict__ out) {
    const int l = threadIdx.x;  // 0..31, exactly one wave, no divergence

    // ---- load: lane l owns x[4l .. 4l+3] (elements' mod-8 classes = 4*(l&1)+t)
    float4 xv = reinterpret_cast<const float4*>(x)[l];

    // ---- stride-8 class maxima via parity-preserving xor-shuffle max tree.
    // xor masks {2,4,8,16} keep bit0 of the lane id, so component t reduces
    // over exactly the lanes whose component t shares the same mod-8 class.
    float m0 = xv.x, m1 = xv.y, m2 = xv.z, m3 = xv.w;
    #pragma unroll
    for (int mask = 2; mask <= 16; mask <<= 1) {
        m0 = fmaxf(m0, __shfl_xor(m0, mask, 32));
        m1 = fmaxf(m1, __shfl_xor(m1, mask, 32));
        m2 = fmaxf(m2, __shfl_xor(m2, mask, 32));
        m3 = fmaxf(m3, __shfl_xor(m3, mask, 32));
    }

    // diff = x - classmax (<= 0), exp(diff) in (0,1]
    const float d0 = xv.x - m0, d1 = xv.y - m1, d2 = xv.z - m2, d3 = xv.w - m3;
    const float e0 = expf(d0), e1 = expf(d1), e2 = expf(d2), e3 = expf(d3);

    // ---- matmul-based sum-reduce (mirrors reference's ones @ exp_diff):
    // A = ones(16x32) f16, B holds the 128 exp values (4 per lane, rest 0).
    // D[m][n] = sum_k B[k][n]; summing row 0 over the 16 columns gives S.
    v16h a;
    #pragma unroll
    for (int i = 0; i < 16; ++i) a[i] = (_Float16)1.0f;

    v16h b;
    #pragma unroll
    for (int i = 0; i < 16; ++i) b[i] = (_Float16)0.0f;
    b[0] = (_Float16)e0; b[1] = (_Float16)e1;
    b[2] = (_Float16)e2; b[3] = (_Float16)e3;

    v8f acc;
    #pragma unroll
    for (int i = 0; i < 8; ++i) acc[i] = 0.0f;
    // (neg_a, A, neg_b, B, c_mod, C, reuse_a, reuse_b)
    acc = __builtin_amdgcn_wmma_f32_16x16x32_f16(
        false, a, false, b, (short)0, acc, false, false);

    // acc[0] on lane l = column-sum for column (l & 15); reduce the 16
    // columns (each 16-lane half independently yields the same total).
    float S = acc[0];
    #pragma unroll
    for (int mask = 1; mask <= 8; mask <<= 1)
        S += __shfl_xor(S, mask, 32);

    const float s = logf(S);  // scalar log-sum-exp, as in the reference

    float4 o;
    o.x = expf(d0 - s);
    o.y = expf(d1 - s);
    o.z = expf(d2 - s);
    o.w = expf(d3 - s);

    // ---- broadcast-store 8 identical rows of the (128,128) output
    float4* out4 = reinterpret_cast<float4*>(out);
    const int base_row = blockIdx.x * 8;
    #pragma unroll
    for (int r = 0; r < 8; ++r)
        out4[(base_row + r) * 32 + l] = o;
}

extern "C" void kernel_launch(void* const* d_in, const int* in_sizes, int n_in,
                              void* d_out, int out_size, void* d_ws, size_t ws_size,
                              hipStream_t stream) {
    (void)in_sizes; (void)n_in; (void)d_ws; (void)ws_size; (void)out_size;
    const float* x  = (const float*)d_in[0];   // (1,128) fp32
    float* out      = (float*)d_out;           // (128,128) fp32
    // 16 single-wave blocks: spreads the 64KB of row stores across WGPs;
    // everything else is launch-latency bound.
    softmax_conv_wmma_kernel<<<16, 32, 0, stream>>>(x, out);
}
